// GResNetBlock_13099650253560
// MI455X (gfx1250) — matrix operand order repaired and checked by reading, not compiled, not run on hardware
//
#include <hip/hip_runtime.h>

// ---------------------------------------------------------------------------
// Types for CDNA5 WMMA (wave32): v_wmma_f32_16x16x32_bf16
// ---------------------------------------------------------------------------
typedef __attribute__((ext_vector_type(16))) __bf16          v16bf;
typedef __attribute__((ext_vector_type(8)))  float           v8f;
typedef __attribute__((ext_vector_type(8)))  unsigned short  v8us;   // 16B LDS vector

union FragCast { v16bf v; v8us h[2]; };

__device__ __forceinline__ unsigned short f2bf(float f) {
  unsigned int u = __float_as_uint(f);
  u += 0x7FFFu + ((u >> 16) & 1u);          // round-to-nearest-even
  return (unsigned short)(u >> 16);
}

// ---------------------------------------------------------------------------
// Utility kernels
// ---------------------------------------------------------------------------
__global__ __launch_bounds__(256) void zero_u32(unsigned int* __restrict__ p, int n) {
  int i = blockIdx.x * 256 + threadIdx.x;
  if (i < n) p[i] = 0u;
}

__global__ __launch_bounds__(256) void f32_to_bf16_k(const float* __restrict__ s,
                                                     unsigned short* __restrict__ d, int n) {
  int i = blockIdx.x * 256 + threadIdx.x;
  if (i < n) d[i] = f2bf(s[i]);
}

__global__ __launch_bounds__(256) void add_vec_k(const float* __restrict__ a,
                                                 const float* __restrict__ b,
                                                 float* __restrict__ c, int n) {
  int i = blockIdx.x * 256 + threadIdx.x;
  if (i < n) c[i] = a[i] + b[i];
}

// ---------------------------------------------------------------------------
// BatchNorm (training mode, biased stats over node dim)
// ---------------------------------------------------------------------------
__global__ __launch_bounds__(256) void bn_stats_k(const float* __restrict__ x,
                                                  float* __restrict__ sum,
                                                  float* __restrict__ sumsq,
                                                  int Nrows, int F, int rowsPerBlock) {
  int r0 = blockIdx.x * rowsPerBlock;
  int r1 = r0 + rowsPerBlock; if (r1 > Nrows) r1 = Nrows;
  for (int c = threadIdx.x; c < F; c += 256) {
    float s = 0.f, q = 0.f;
    for (int r = r0; r < r1; ++r) {
      float v = x[(size_t)r * F + c];
      s += v; q += v * v;
    }
    atomicAdd(&sum[c], s);
    atomicAdd(&sumsq[c], q);
  }
}

__global__ __launch_bounds__(256) void bn_finalize_k(const float* __restrict__ sum,
                                                     const float* __restrict__ sumsq,
                                                     float* __restrict__ mean,
                                                     float* __restrict__ rstd,
                                                     int Nrows, int F) {
  int c = blockIdx.x * 256 + threadIdx.x;
  if (c < F) {
    float inv = 1.0f / (float)Nrows;
    float mu  = sum[c] * inv;
    float var = sumsq[c] * inv - mu * mu;
    mean[c] = mu;
    rstd[c] = rsqrtf(var + 1e-5f);
  }
}

// h = relu((x-mu)*rstd*gamma+beta); write f32 copy (for edge gather) + bf16 copy (for WMMA)
__global__ __launch_bounds__(256) void bn_apply_relu_k(const float* __restrict__ x,
                                                       const float* __restrict__ mean,
                                                       const float* __restrict__ rstd,
                                                       const float* __restrict__ gamma,
                                                       const float* __restrict__ beta,
                                                       float* __restrict__ hf,
                                                       unsigned short* __restrict__ hb,
                                                       int total, int Fmask) {
  int i = blockIdx.x * 256 + threadIdx.x;
  if (i < total) {
    int c = i & Fmask;
    float v = (x[i] - mean[c]) * rstd[c] * gamma[c] + beta[c];
    v = v > 0.f ? v : 0.f;
    hf[i] = v;
    hb[i] = f2bf(v);
  }
}

// ---------------------------------------------------------------------------
// Graph aggregation
// ---------------------------------------------------------------------------
__global__ __launch_bounds__(256) void edge_count_k(const int* __restrict__ dst,
                                                    int* __restrict__ cnt, int E) {
  int e = blockIdx.x * 256 + threadIdx.x;
  if (e < E) atomicAdd(&cnt[dst[e]], 1);
}

// 64 lanes per edge; float4 gather of source row, f32 atomic scatter to dest row
__global__ __launch_bounds__(256) void scatter_add_k(const float* __restrict__ h,
                                                     const int* __restrict__ src,
                                                     const int* __restrict__ dst,
                                                     float* __restrict__ agg,
                                                     int F, int E) {
  int e = blockIdx.x * 4 + (threadIdx.x >> 6);
  if (e >= E) return;
  int sub = threadIdx.x & 63;
  int s = src[e], d = dst[e];
  const float* hp = h + (size_t)s * F;
  float* ap = agg + (size_t)d * F;
  for (int c4 = sub * 4; c4 < F; c4 += 256) {
    float4 v = *(const float4*)(hp + c4);
    atomicAdd(ap + c4 + 0, v.x);
    atomicAdd(ap + c4 + 1, v.y);
    atomicAdd(ap + c4 + 2, v.z);
    atomicAdd(ap + c4 + 3, v.w);
  }
}

__global__ __launch_bounds__(256) void mean_to_bf_k(const float* __restrict__ agg,
                                                    const int* __restrict__ cnt,
                                                    unsigned short* __restrict__ mb,
                                                    int total, int logF) {
  int i = blockIdx.x * 256 + threadIdx.x;
  if (i < total) {
    int n = i >> logF;
    int c = cnt[n];
    float cf = c > 0 ? (float)c : 1.0f;
    mb[i] = f2bf(agg[i] / cf);
  }
}

// ---------------------------------------------------------------------------
// Multi-source WMMA GEMM:  C = sum_p A_p @ B_p + bias   (A bf16 [M,Kp], B bf16 [Kp,Nout])
// Block tile 128x64, 8 waves, each wave: one 16-row strip x 64 cols (4 wmma tiles).
// A panel row-major (rows padded to 40 halfs -> 80B, 16B-aligned chunks);
// B panel stored TRANSPOSED in LDS so each fragment is contiguous in K:
//   A frag = 2 x ds_load_b128, B frag = 2 x ds_load_b128.
// ---------------------------------------------------------------------------
#define BM 128
#define BNT 64
#define BK 32
#define APAD 40   // row stride (halfs) for A panel: 80B, multiple of 16B
#define BPAD 40   // row stride (halfs) for transposed B panel

__global__ __launch_bounds__(256) void gemm_bf16_multi_k(
    const unsigned short* __restrict__ A0, const unsigned short* __restrict__ B0, int K0,
    const unsigned short* __restrict__ A1, const unsigned short* __restrict__ B1, int K1,
    const unsigned short* __restrict__ A2, const unsigned short* __restrict__ B2, int K2,
    const float* __restrict__ bias, float* __restrict__ C, int M, int Nout) {

  __shared__ __align__(16) unsigned short As[BM][APAD];    // [row][k]
  __shared__ __align__(16) unsigned short Bst[BNT][BPAD];  // [col][k]  (transposed)

  const int tid    = threadIdx.x;
  const int lane   = tid & 31;
  const int wave   = tid >> 5;                // 0..7 (wave32)
  const int blockM = blockIdx.x * BM;
  const int blockN = blockIdx.y * BNT;
  const int laneHi = (lane >> 4) & 1;
  const int lm     = lane & 15;

  v8f acc[4];
  v8f z;
#pragma unroll
  for (int j = 0; j < 8; ++j) z[j] = 0.f;
#pragma unroll
  for (int t = 0; t < 4; ++t) acc[t] = z;

  const unsigned short* Aps[3] = {A0, A1, A2};
  const unsigned short* Bps[3] = {B0, B1, B2};
  const int Ks[3] = {K0, K1, K2};

  const int mrow = wave * 16 + lm;            // A row within tile for this lane

  for (int p = 0; p < 3; ++p) {
    const unsigned short* A = Aps[p];
    if (A == nullptr) continue;
    const unsigned short* B = Bps[p];
    const int K = Ks[p];

    for (int k0 = 0; k0 < K; k0 += BK) {
      // Stage A panel [128 x 32] bf16 (coalesced dword loads, zero-fill rows >= M)
      for (int idx = tid; idx < BM * (BK / 2); idx += 256) {
        int r  = idx >> 4;
        int kk = (idx & 15) << 1;
        int gr = blockM + r;
        unsigned int v = 0u;
        if (gr < M) v = *(const unsigned int*)(A + (size_t)gr * K + k0 + kk);
        *(unsigned int*)&As[r][kk] = v;
      }
      // Stage B panel [32 x 64] bf16: coalesced dword global reads, transposed LDS store
      for (int idx = tid; idx < BK * (BNT / 2); idx += 256) {
        int r  = idx >> 5;                    // k row 0..31
        int cc = (idx & 31) << 1;             // column pair
        unsigned int w = *(const unsigned int*)(B + (size_t)(k0 + r) * Nout + blockN + cc);
        Bst[cc + 0][r] = (unsigned short)(w & 0xffffu);
        Bst[cc + 1][r] = (unsigned short)(w >> 16);
      }
      __syncthreads();

      // A fragment: halfs 0-7 -> K = 8*laneHi + [0,8), halfs 8-15 -> K = 16 + 8*laneHi + [0,8)
      FragCast fa;
      fa.h[0] = *(const v8us*)&As[mrow][laneHi * 8];
      fa.h[1] = *(const v8us*)&As[mrow][16 + laneHi * 8];

#pragma unroll
      for (int t = 0; t < 4; ++t) {
        // B fragment: halfs 0-7 -> K = 8*laneHi + [0,8), halfs 8-15 -> K = 16 + 8*laneHi + [0,8) at fixed column
        FragCast fb;
        const unsigned short* bp = &Bst[t * 16 + lm][laneHi * 8];
        fb.h[0] = *(const v8us*)(bp + 0);
        fb.h[1] = *(const v8us*)(bp + 16);
        acc[t] = __builtin_amdgcn_wmma_f32_16x16x32_bf16(
            false, fa.v, false, fb.v, (short)0, acc[t], false, false);
      }
      __syncthreads();
    }
  }

  // Epilogue: C/D f32 16x16 layout — vgpr v -> M = v + 8*laneHi, N = lane&15
#pragma unroll
  for (int t = 0; t < 4; ++t) {
    int col = blockN + t * 16 + lm;
    float bv = bias ? bias[col] : 0.f;
#pragma unroll
    for (int v = 0; v < 8; ++v) {
      int row = blockM + wave * 16 + v + (laneHi << 3);
      if (row < M) C[(size_t)row * Nout + col] = acc[t][v] + bv;
    }
  }
}

// ---------------------------------------------------------------------------
// Host-side orchestration
// ---------------------------------------------------------------------------
extern "C" void kernel_launch(void* const* d_in, const int* in_sizes, int n_in,
                              void* d_out, int out_size, void* d_ws, size_t ws_size,
                              hipStream_t stream) {
  const int FI = 256, FO = 512;
  const int N = in_sizes[0] / FI;
  const int E = in_sizes[1] / 2;

  const float* x      = (const float*)d_in[0];
  const int*   eidx   = (const int*)d_in[1];
  const int*   src    = eidx;
  const int*   dst    = eidx + E;
  const float* gamma0 = (const float*)d_in[2];
  const float* beta0  = (const float*)d_in[3];
  const float* W0l    = (const float*)d_in[4];
  const float* W0r    = (const float*)d_in[5];
  const float* bias0  = (const float*)d_in[6];
  const float* gamma1 = (const float*)d_in[7];
  const float* beta1  = (const float*)d_in[8];
  const float* W1l    = (const float*)d_in[9];
  const float* W1r    = (const float*)d_in[10];
  const float* bias1  = (const float*)d_in[11];
  const float* Wsc    = (const float*)d_in[12];
  const float* bsc    = (const float*)d_in[13];

  char* wsb = (char*)d_ws;
  size_t off = 0;
  auto carve = [&](size_t bytes) -> char* {
    char* p = wsb + off;
    off = (off + bytes + 255) & ~(size_t)255;
    return p;
  };

  unsigned short* W0lb = (unsigned short*)carve((size_t)FI * FO * 2);
  unsigned short* W0rb = (unsigned short*)carve((size_t)FI * FO * 2);
  unsigned short* W1lb = (unsigned short*)carve((size_t)FO * FO * 2);
  unsigned short* W1rb = (unsigned short*)carve((size_t)FO * FO * 2);
  unsigned short* Wscb = (unsigned short*)carve((size_t)FI * FO * 2);
  unsigned short* xb   = (unsigned short*)carve((size_t)N * FI * 2);
  float*          h0f  = (float*)carve((size_t)N * FI * 4);
  unsigned short* h0b  = (unsigned short*)carve((size_t)N * FI * 2);
  float*          h1f  = (float*)carve((size_t)N * FO * 4);
  unsigned short* h1b  = (unsigned short*)carve((size_t)N * FO * 2);
  float*          aggf = (float*)carve((size_t)N * FO * 4);
  unsigned short* mnb  = (unsigned short*)carve((size_t)N * FO * 2);
  float*          out0 = (float*)carve((size_t)N * FO * 4);
  int*            cnt  = (int*)carve((size_t)N * 4);
  float*          sumv = (float*)carve((size_t)FO * 4);
  float*          sqv  = (float*)carve((size_t)FO * 4);
  float*          mnv  = (float*)carve((size_t)FO * 4);
  float*          rsv  = (float*)carve((size_t)FO * 4);
  float*          bC   = (float*)carve((size_t)FO * 4);
  (void)ws_size; (void)n_in;

  auto cdiv = [](int a, int b) { return (a + b - 1) / b; };
  const int rowsPB = cdiv(N, 128);

  // --- one-time per-call conversions (weights/x to bf16, combined bias) ---
  f32_to_bf16_k<<<cdiv(FI * FO, 256), 256, 0, stream>>>(W0l, W0lb, FI * FO);
  f32_to_bf16_k<<<cdiv(FI * FO, 256), 256, 0, stream>>>(W0r, W0rb, FI * FO);
  f32_to_bf16_k<<<cdiv(FO * FO, 256), 256, 0, stream>>>(W1l, W1lb, FO * FO);
  f32_to_bf16_k<<<cdiv(FO * FO, 256), 256, 0, stream>>>(W1r, W1rb, FO * FO);
  f32_to_bf16_k<<<cdiv(FI * FO, 256), 256, 0, stream>>>(Wsc, Wscb, FI * FO);
  f32_to_bf16_k<<<cdiv(N * FI, 256), 256, 0, stream>>>(x, xb, N * FI);
  add_vec_k<<<cdiv(FO, 256), 256, 0, stream>>>(bias1, bsc, bC, FO);

  // --- in-degree counts ---
  zero_u32<<<cdiv(N, 256), 256, 0, stream>>>((unsigned int*)cnt, N);
  edge_count_k<<<cdiv(E, 256), 256, 0, stream>>>(dst, cnt, E);

  // --- layer 0: BN + ReLU ---
  zero_u32<<<cdiv(FI, 256), 256, 0, stream>>>((unsigned int*)sumv, FI);
  zero_u32<<<cdiv(FI, 256), 256, 0, stream>>>((unsigned int*)sqv, FI);
  bn_stats_k<<<128, 256, 0, stream>>>(x, sumv, sqv, N, FI, rowsPB);
  bn_finalize_k<<<cdiv(FI, 256), 256, 0, stream>>>(sumv, sqv, mnv, rsv, N, FI);
  bn_apply_relu_k<<<cdiv(N * FI, 256), 256, 0, stream>>>(x, mnv, rsv, gamma0, beta0,
                                                         h0f, h0b, N * FI, FI - 1);

  // --- layer 0: mean aggregation ---
  zero_u32<<<cdiv(N * FI, 256), 256, 0, stream>>>((unsigned int*)aggf, N * FI);
  scatter_add_k<<<cdiv(E, 4), 256, 0, stream>>>(h0f, src, dst, aggf, FI, E);
  mean_to_bf_k<<<cdiv(N * FI, 256), 256, 0, stream>>>(aggf, cnt, mnb, N * FI, 8);

  // --- layer 0 GEMM: out0 = mean0@W0l + h0@W0r + bias0 ---
  gemm_bf16_multi_k<<<dim3(cdiv(N, BM), FO / BNT), 256, 0, stream>>>(
      mnb, W0lb, FI, h0b, W0rb, FI, nullptr, nullptr, 0, bias0, out0, N, FO);

  // --- layer 1: BN + ReLU ---
  zero_u32<<<cdiv(FO, 256), 256, 0, stream>>>((unsigned int*)sumv, FO);
  zero_u32<<<cdiv(FO, 256), 256, 0, stream>>>((unsigned int*)sqv, FO);
  bn_stats_k<<<128, 256, 0, stream>>>(out0, sumv, sqv, N, FO, rowsPB);
  bn_finalize_k<<<cdiv(FO, 256), 256, 0, stream>>>(sumv, sqv, mnv, rsv, N, FO);
  bn_apply_relu_k<<<cdiv(N * FO, 256), 256, 0, stream>>>(out0, mnv, rsv, gamma1, beta1,
                                                         h1f, h1b, N * FO, FO - 1);

  // --- layer 1: mean aggregation ---
  zero_u32<<<cdiv(N * FO, 256), 256, 0, stream>>>((unsigned int*)aggf, N * FO);
  scatter_add_k<<<cdiv(E, 4), 256, 0, stream>>>(h1f, src, dst, aggf, FO, E);
  mean_to_bf_k<<<cdiv(N * FO, 256), 256, 0, stream>>>(aggf, cnt, mnb, N * FO, 9);

  // --- final fused GEMM: d_out = mean1@W1l + h1@W1r + x@Wsc + (bias1+b_sc) ---
  gemm_bf16_multi_k<<<dim3(cdiv(N, BM), FO / BNT), 256, 0, stream>>>(
      mnb, W1lb, FO, h1b, W1rb, FO, xb, Wscb, FI, bC, (float*)d_out, N, FO);
}
